// eGAUSSp_62173946577545
// MI455X (gfx1250) — compile-verified, matrix operand order checked by reading, not run on hardware
//
#include <hip/hip_runtime.h>

// ---------------------------------------------------------------------------
// eGAUSSp inference on gfx1250 (MI455X), wave32 + WMMA bf16 split-precision.
// Phases:
//   1) invert_sigma_kernel : Sinv_c = (S_c/n_c + eps I)^-1, bf16 hi/lo to ws
//   2) mahal_wmma_kernel   : d2[n,c] via v_wmma_f32_16x16x32_bf16 (3-way split)
//                            + async global->LDS staging + ds_swizzle reduce
//   3) score_kernel        : scores / pred / clusters reductions
// ---------------------------------------------------------------------------

typedef __attribute__((ext_vector_type(16))) __bf16 v16bf;
typedef __attribute__((ext_vector_type(8)))  __bf16 bf16x8;
typedef __attribute__((ext_vector_type(8)))  float  v8f;
typedef int v4i_vs __attribute__((vector_size(16)));   // async-to-LDS operand

#define N_SAMPLES 4096
#define C_CLUST   512
#define D_DIM     64
#define K_CLASS   10

// xor-butterfly sum over 16 lanes via ds_swizzle (imm xor masks 1,2,4,8;
// masks < 16 keep the two 16-lane halves of the wave independent).
__device__ __forceinline__ float xor_reduce16(float p) {
  p += __int_as_float(__builtin_amdgcn_ds_swizzle(__float_as_int(p), 0x041F));
  p += __int_as_float(__builtin_amdgcn_ds_swizzle(__float_as_int(p), 0x081F));
  p += __int_as_float(__builtin_amdgcn_ds_swizzle(__float_as_int(p), 0x101F));
  p += __int_as_float(__builtin_amdgcn_ds_swizzle(__float_as_int(p), 0x201F));
  return p;
}

// ---------------------------------------------------------------------------
// Phase 1: per-cluster SPD inversion via Gauss-Jordan on [Sigma | I] in LDS.
// ---------------------------------------------------------------------------
__global__ void __launch_bounds__(256) invert_sigma_kernel(
    const float* __restrict__ S,      // [C,D,D]
    const float* __restrict__ nvec,   // [C]
    __bf16* __restrict__ sinv_hi,     // [C,D,D]
    __bf16* __restrict__ sinv_lo)     // [C,D,D]
{
  __shared__ float M[64][130];        // padded augmented matrix [Sigma | I]
  __shared__ float colp[64];
  __shared__ float invp;

  const int c   = blockIdx.x;
  const int tid = threadIdx.x;
  const float inv_n = 1.0f / nvec[c];

  for (int idx = tid; idx < 64 * 64; idx += 256) {
    const int r = idx >> 6, j = idx & 63;
    float v = S[c * 4096 + idx] * inv_n + ((r == j) ? 1e-6f : 0.0f);
    M[r][j]      = v;
    M[r][64 + j] = (r == j) ? 1.0f : 0.0f;
  }
  __syncthreads();

  for (int p = 0; p < 64; ++p) {
    if (tid == 0) invp = 1.0f / M[p][p];
    __syncthreads();
    if (tid < 128) {
      M[p][tid] *= invp;                       // normalize pivot row
    } else if (tid < 192) {
      const int r = tid - 128;
      if (r != p) colp[r] = M[r][p];           // stash pivot column
    }
    __syncthreads();
    for (int idx = tid; idx < 64 * 128; idx += 256) {
      const int r = idx >> 7, col = idx & 127;
      if (r != p) M[r][col] -= colp[r] * M[p][col];
    }
    __syncthreads();
  }

  for (int idx = tid; idx < 64 * 64; idx += 256) {
    const int r = idx >> 6, j = idx & 63;
    const float v = M[r][64 + j];
    const __bf16 h = (__bf16)v;
    sinv_hi[c * 4096 + idx] = h;
    sinv_lo[c * 4096 + idx] = (__bf16)(v - (float)h);
  }
}

// ---------------------------------------------------------------------------
// Phase 2: squared Mahalanobis via WMMA.
// Block = 256 thr (8 waves); grid = (N/64, C). Y = diff * Sinv (Sinv symmetric
// => B-fragment rows == columns, contiguous-K 16B LDS loads for both A & B).
// Split product: Y ~= Ah*Bh + Ah*Bl + Al*Bh (f32 acc). Two output tiles per
// wave share A fragments; their WMMA chains are interleaved to hide the
// WMMA->WMMA RAW hazard slots.
// ---------------------------------------------------------------------------
__global__ void __launch_bounds__(256) mahal_wmma_kernel(
    const float*  __restrict__ data,     // [N,D]
    const float*  __restrict__ mu,       // [C,D]
    const __bf16* __restrict__ sinv_hi,  // [C,D,D]
    const __bf16* __restrict__ sinv_lo,  // [C,D,D]
    float* __restrict__ gamma)           // [N,C]
{
  __shared__ __bf16 sA_hi[64][64];
  __shared__ __bf16 sA_lo[64][64];
  __shared__ __bf16 sB_hi[64][64];
  __shared__ __bf16 sB_lo[64][64];
  __shared__ float  sDiff[64][65];       // padded: conflict-free column reads
  __shared__ float  sD2[64];

  const int c   = blockIdx.y;
  const int m0  = blockIdx.x * 64;
  const int tid = threadIdx.x;

  // Stage Sinv hi/lo via async global->LDS DMA (ASYNCcnt path) if available.
#if __has_builtin(__builtin_amdgcn_global_load_async_to_lds_b128)
  {
    const char* ghi = (const char*)(sinv_hi + (size_t)c * 4096);
    const char* glo = (const char*)(sinv_lo + (size_t)c * 4096);
    char* lhi = (char*)&sB_hi[0][0];
    char* llo = (char*)&sB_lo[0][0];
    #pragma unroll
    for (int it = 0; it < 2; ++it) {
      const int seg = (tid + it * 256) * 16;       // 512 x 16B per matrix
      __builtin_amdgcn_global_load_async_to_lds_b128(
          (v4i_vs*)(ghi + seg), (v4i_vs*)(lhi + seg), 0, 0);
      __builtin_amdgcn_global_load_async_to_lds_b128(
          (v4i_vs*)(glo + seg), (v4i_vs*)(llo + seg), 0, 0);
    }
  }
#else
  for (int idx = tid; idx < 64 * 64; idx += 256) {
    (&sB_hi[0][0])[idx] = sinv_hi[c * 4096 + idx];
    (&sB_lo[0][0])[idx] = sinv_lo[c * 4096 + idx];
  }
#endif

  // Build diff tile (f32 + bf16 hi/lo split).
  for (int idx = tid; idx < 64 * 64; idx += 256) {
    const int r = idx >> 6, k = idx & 63;
    const float v = data[(m0 + r) * 64 + k] - mu[c * 64 + k];
    sDiff[r][k] = v;
    const __bf16 h = (__bf16)v;
    sA_hi[r][k] = h;
    sA_lo[r][k] = (__bf16)(v - (float)h);
  }
  if (tid < 64) sD2[tid] = 0.0f;

#if __has_builtin(__builtin_amdgcn_global_load_async_to_lds_b128)
#if __has_builtin(__builtin_amdgcn_s_wait_asynccnt)
  __builtin_amdgcn_s_wait_asynccnt(0);
#else
  asm volatile("s_wait_asynccnt 0x0" ::: "memory");
#endif
#endif
  __syncthreads();

  const int wave = tid >> 5;
  const int lane = tid & 31;
  const int lrow = lane & 15;   // A: row M; B: column N (== row, symmetric)
  const int lhi  = lane >> 4;   // K half select (+8)
  const int mt   = wave & 3;
  const int nt0  = wave >> 2;   // tile pair shares A fragments
  const int nt1  = nt0 + 2;

  v8f acc0 = {}, acc1 = {};
  #pragma unroll
  for (int kc = 0; kc < 2; ++kc) {
    v16bf a_hi, a_lo, b0_hi, b0_lo, b1_hi, b1_lo;
    #pragma unroll
    for (int g = 0; g < 2; ++g) {
      // v16bf elements [8g..8g+7] hold 8 consecutive K values
      const int koff = kc * 32 + g * 16 + lhi * 8;
      const bf16x8 vah  = *(const bf16x8*)&sA_hi[mt  * 16 + lrow][koff];
      const bf16x8 val  = *(const bf16x8*)&sA_lo[mt  * 16 + lrow][koff];
      const bf16x8 vb0h = *(const bf16x8*)&sB_hi[nt0 * 16 + lrow][koff];
      const bf16x8 vb0l = *(const bf16x8*)&sB_lo[nt0 * 16 + lrow][koff];
      const bf16x8 vb1h = *(const bf16x8*)&sB_hi[nt1 * 16 + lrow][koff];
      const bf16x8 vb1l = *(const bf16x8*)&sB_lo[nt1 * 16 + lrow][koff];
      #pragma unroll
      for (int t = 0; t < 8; ++t) {
        a_hi[g * 8 + t]  = vah[t];
        a_lo[g * 8 + t]  = val[t];
        b0_hi[g * 8 + t] = vb0h[t];
        b0_lo[g * 8 + t] = vb0l[t];
        b1_hi[g * 8 + t] = vb1h[t];
        b1_lo[g * 8 + t] = vb1l[t];
      }
    }
    // Interleave the two independent accumulator chains.
    acc0 = __builtin_amdgcn_wmma_f32_16x16x32_bf16(false, a_hi, false, b0_hi,
                                                   (short)0, acc0, false, false);
    acc1 = __builtin_amdgcn_wmma_f32_16x16x32_bf16(false, a_hi, false, b1_hi,
                                                   (short)0, acc1, false, false);
    acc0 = __builtin_amdgcn_wmma_f32_16x16x32_bf16(false, a_hi, false, b0_lo,
                                                   (short)0, acc0, false, false);
    acc1 = __builtin_amdgcn_wmma_f32_16x16x32_bf16(false, a_hi, false, b1_lo,
                                                   (short)0, acc1, false, false);
    acc0 = __builtin_amdgcn_wmma_f32_16x16x32_bf16(false, a_lo, false, b0_hi,
                                                   (short)0, acc0, false, false);
    acc1 = __builtin_amdgcn_wmma_f32_16x16x32_bf16(false, a_lo, false, b1_hi,
                                                   (short)0, acc1, false, false);
  }

  // d2 partial: sum_e Y[m,e]*diff[m,e]; ds_swizzle xor-reduce over 16 N-lanes.
  #pragma unroll
  for (int i = 0; i < 8; ++i) {
    const int m_local = mt * 16 + i + lhi * 8;
    const float p0 = acc0[i] * sDiff[m_local][nt0 * 16 + lrow];
    const float p1 = acc1[i] * sDiff[m_local][nt1 * 16 + lrow];
    const float r  = xor_reduce16(p0) + xor_reduce16(p1);
    if (lrow == 0) atomicAdd(&sD2[m_local], r);
  }
  __syncthreads();

  if (tid < 64) {
    const float g = __expf(-0.5f * sD2[tid]);
    gamma[(size_t)(m0 + tid) * C_CLUST + c] = g;
  }
}

// ---------------------------------------------------------------------------
// Phase 3: per-sample reduction over clusters.
// scores = (Gamma @ onehot)/sum(Gamma); pred = argmax scores; clusters =
// argmax Gamma (first-max tiebreak via packed key, ds_max_u64 in LDS).
// ---------------------------------------------------------------------------
__global__ void __launch_bounds__(256) score_kernel(
    const float* __restrict__ gamma,          // [N,C]
    const int*   __restrict__ cluster_labels, // [C,K] one-hot
    float* __restrict__ out_scores,           // [N,K]
    float* __restrict__ out_pred,             // [N]
    float* __restrict__ out_clusters)         // [N]
{
  __shared__ float csum[K_CLASS];
  __shared__ float tot;
  __shared__ unsigned long long best;

  const int nidx = blockIdx.x;
  const int tid  = threadIdx.x;
  if (tid < K_CLASS) csum[tid] = 0.0f;
  if (tid == 0) { tot = 0.0f; best = 0ULL; }
  __syncthreads();

  float lsum = 0.0f;
  unsigned long long lbest = 0ULL;
  #pragma unroll
  for (int it = 0; it < 2; ++it) {
    const int c = tid + it * 256;
    const float g = gamma[(size_t)nidx * C_CLUST + c];
    lsum += g;
    // Gamma > 0 => float bits monotone as uint; low word = ~c so that
    // on equal Gamma the SMALLEST cluster index wins (matches jnp.argmax).
    const unsigned long long key =
        ((unsigned long long)__float_as_uint(g) << 32) |
        (unsigned long long)(0xFFFFFFFFu - (unsigned)c);
    if (key > lbest) lbest = key;
    int cls = 0;
    #pragma unroll
    for (int k = 0; k < K_CLASS; ++k)
      if (cluster_labels[c * K_CLASS + k] != 0) cls = k;
    atomicAdd(&csum[cls], g);
  }
  atomicAdd(&tot, lsum);
  atomicMax(&best, lbest);
  __syncthreads();

  if (tid < K_CLASS)
    out_scores[nidx * K_CLASS + tid] = csum[tid] / (tot + 1e-12f);
  if (tid == 0) {
    const float denom = tot + 1e-12f;
    int pk = 0;
    float pv = csum[0] / denom;
    #pragma unroll
    for (int k = 1; k < K_CLASS; ++k) {
      const float v = csum[k] / denom;
      if (v > pv) { pv = v; pk = k; }   // strict > => first max
    }
    out_pred[nidx]     = (float)pk;
    out_clusters[nidx] = (float)(0xFFFFFFFFu - (unsigned)(best & 0xFFFFFFFFu));
  }
}

// ---------------------------------------------------------------------------
extern "C" void kernel_launch(void* const* d_in, const int* in_sizes, int n_in,
                              void* d_out, int out_size, void* d_ws, size_t ws_size,
                              hipStream_t stream) {
  const float* data           = (const float*)d_in[0];   // [N,D]
  // d_in[1] = labels (unused by the inference path, matches reference)
  const float* nvec           = (const float*)d_in[2];   // [C]
  const float* mu             = (const float*)d_in[3];   // [C,D]
  const float* S              = (const float*)d_in[4];   // [C,D,D]
  const int*   cluster_labels = (const int*)d_in[5];     // [C,K]

  // Workspace layout: Sinv hi (4MB bf16) | Sinv lo (4MB) | Gamma (8MB f32)
  __bf16* sinv_hi = (__bf16*)d_ws;
  __bf16* sinv_lo = sinv_hi + (size_t)C_CLUST * D_DIM * D_DIM;
  float*  gamma   = (float*)((char*)d_ws +
                             2ull * C_CLUST * D_DIM * D_DIM * sizeof(__bf16));

  float* out_scores   = (float*)d_out;
  float* out_pred     = out_scores + (size_t)N_SAMPLES * K_CLASS;
  float* out_clusters = out_pred + N_SAMPLES;

  invert_sigma_kernel<<<C_CLUST, 256, 0, stream>>>(S, nvec, sinv_hi, sinv_lo);
  mahal_wmma_kernel<<<dim3(N_SAMPLES / 64, C_CLUST), 256, 0, stream>>>(
      data, mu, sinv_hi, sinv_lo, gamma);
  score_kernel<<<N_SAMPLES, 256, 0, stream>>>(gamma, cluster_labels,
                                              out_scores, out_pred, out_clusters);
}
